// MultiHeadAttention_14293651161630
// MI455X (gfx1250) — matrix operand order repaired and checked by reading, not compiled, not run on hardware
//
#include <hip/hip_runtime.h>
#include <hip/hip_bf16.h>
#include <cstdint>

typedef __attribute__((ext_vector_type(16))) _Float16 v16h;
typedef __attribute__((ext_vector_type(8)))  float    v8f;

#define TILE_K     32
#define BM         128
#define BN         128
#define LDSTRIDE   48   // halfs per LDS row: 32 data + 16 pad (96B, 16B aligned)

// ---------------------------------------------------------------------------
// Stage 1: Out[m,n] = (f16) ( sum_k X[m,k] * W[n,k] + bias[n] )
// X: [M,K] f32, W: [N,K] f32 (torch Linear weight), Out: [M,N] f16
// Block tile 128x128, 8 waves, wave tile 64x32 (4x2 subtiles of 16x16).
// Double-buffered LDS; next tile's global loads issued before the WMMAs.
// ---------------------------------------------------------------------------
__global__ void __launch_bounds__(256)
qkv_gemm(const float* __restrict__ X, const float* __restrict__ W,
         const float* __restrict__ bias, _Float16* __restrict__ Out,
         int M, int N, int K)
{
    __shared__ _Float16 lA[2][BM * LDSTRIDE];   // 2 x 12 KB
    __shared__ _Float16 lB[2][BN * LDSTRIDE];   // 2 x 12 KB

    const int tid  = threadIdx.x;
    const int lane = tid & 31;
    const int wave = tid >> 5;        // 0..7
    const int lr   = lane & 15;       // lane-in-half
    const int hi   = lane >> 4;       // which 16-lane half
    const int wm   = wave >> 2;       // 0..1 : 64-row band
    const int wn   = wave & 3;        // 0..3 : 32-col band

    const int m0 = blockIdx.y * BM;
    const int n0 = blockIdx.x * BN;

    float4 ra[4], rb[4];              // staged global data (regs)

    auto gload = [&](int k0) {        // issue all 8 b128 loads back-to-back
        #pragma unroll
        for (int j = 0; j < 4; ++j) {
            const int c = tid + j * 256;
            ra[j] = *(const float4*)(X + (size_t)(m0 + (c >> 3)) * K + k0 + ((c & 7) << 2));
        }
        #pragma unroll
        for (int j = 0; j < 4; ++j) {
            const int c = tid + j * 256;
            rb[j] = *(const float4*)(W + (size_t)(n0 + (c >> 3)) * K + k0 + ((c & 7) << 2));
        }
    };
    auto lstore = [&](int buf) {      // f32 -> f16 convert + LDS store
        #pragma unroll
        for (int j = 0; j < 4; ++j) {
            const int c = tid + j * 256;
            _Float16* d = &lA[buf][(c >> 3) * LDSTRIDE + ((c & 7) << 2)];
            d[0] = (_Float16)ra[j].x; d[1] = (_Float16)ra[j].y;
            d[2] = (_Float16)ra[j].z; d[3] = (_Float16)ra[j].w;
        }
        #pragma unroll
        for (int j = 0; j < 4; ++j) {
            const int c = tid + j * 256;
            _Float16* d = &lB[buf][(c >> 3) * LDSTRIDE + ((c & 7) << 2)];
            d[0] = (_Float16)rb[j].x; d[1] = (_Float16)rb[j].y;
            d[2] = (_Float16)rb[j].z; d[3] = (_Float16)rb[j].w;
        }
    };

    gload(0);
    lstore(0);
    int buf = 0;

    v8f acc[4][2] = {};               // [m-subtile][n-subtile]

    for (int k0 = 0; k0 < K; k0 += TILE_K) {
        __syncthreads();              // LDS[buf] tile visible to all waves
        const bool more = (k0 + TILE_K) < K;
        if (more) {
            gload(k0 + TILE_K);       // overlap with WMMAs below
            if (k0 + 2 * TILE_K < K)  // L2 prefetch of the tile after next
                __builtin_prefetch(X + (size_t)(m0 + (tid >> 1)) * K
                                     + k0 + 2 * TILE_K + ((tid & 1) << 4), 0, 0);
        }

        // ---- gather fragments per documented CDNA5 VGPR layouts ----
        v16h afrag[4], bfrag[2];
        #pragma unroll
        for (int mt = 0; mt < 4; ++mt) {
            const int row = wm * 64 + mt * 16 + lr;          // A row M = lane%16
            const uint32_t* src = (const uint32_t*)&lA[buf][row * LDSTRIDE];
            uint32_t* ap = (uint32_t*)&afrag[mt];
            #pragma unroll
            for (int p = 0; p < 8; ++p) {
                const int kk = (p < 4) ? (hi * 8 + 2 * p) : (16 + hi * 8 + 2 * (p - 4));
                ap[p] = src[kk >> 1];
            }
        }
        #pragma unroll
        for (int nt = 0; nt < 2; ++nt) {
            const int row = wn * 32 + nt * 16 + lr;          // B col N = lane%16
            const uint32_t* src = (const uint32_t*)&lB[buf][row * LDSTRIDE];
            uint32_t* bp = (uint32_t*)&bfrag[nt];
            #pragma unroll
            for (int p = 0; p < 8; ++p)
                bp[p] = src[((p < 4) ? (hi * 8 + 2 * p) : (16 + hi * 8 + 2 * (p - 4))) >> 1]; // K = table order, same as A
        }

        #pragma unroll
        for (int mt = 0; mt < 4; ++mt)
            #pragma unroll
            for (int nt = 0; nt < 2; ++nt)
                acc[mt][nt] = __builtin_amdgcn_wmma_f32_16x16x32_f16(
                    false, afrag[mt], false, bfrag[nt],
                    (short)0, acc[mt][nt], false, false);

        if (more) lstore(buf ^ 1);    // safe: other buffer
        buf ^= 1;
    }

    // ---- epilogue: +bias, convert to f16, store ----
    #pragma unroll
    for (int mt = 0; mt < 4; ++mt) {
        #pragma unroll
        for (int nt = 0; nt < 2; ++nt) {
            const int Ni = n0 + wn * 32 + nt * 16 + lr;
            const float b = bias[Ni];
            #pragma unroll
            for (int i = 0; i < 8; ++i) {
                const int Mi = m0 + wm * 64 + mt * 16 + hi * 8 + i;
                Out[(size_t)Mi * N + Ni] = (_Float16)(acc[mt][nt][i] + b);
            }
        }
    }
}

// ---------------------------------------------------------------------------
// Stage 2: per token (32 heads x 32 dh):
//   scores[h,t] = sum_d q[h,d] k[t,d] / 32 ; softmax over t ; out = scores*v
// One wave per token, 8 tokens per block. 4 WMMAs per wave for scores.
// Staging via CDNA5 async global->LDS copies (ASYNCcnt). Dynamic LDS so the
// async destination address is a plain byte offset (base 0, no static LDS).
// ---------------------------------------------------------------------------
__global__ void __launch_bounds__(256)
attn_stage2(const _Float16* __restrict__ Q, const _Float16* __restrict__ Kp,
            const _Float16* __restrict__ V, float* __restrict__ Out)
{
    extern __shared__ _Float16 smem[];  // 48 KB: [0)Q [16K)K [32K)V

    const int tid  = threadIdx.x;
    const int lane = tid & 31;
    const int w    = tid >> 5;          // wave -> token within block
    const int lr   = lane & 15;
    const int hi   = lane >> 4;
    const int t0   = blockIdx.x * 8;

    // ---- async-stage 8 tokens (16KB per tensor) into LDS ----
    {
        const uint64_t gq = (uint64_t)(Q  + (size_t)t0 * 1024);
        const uint64_t gk = (uint64_t)(Kp + (size_t)t0 * 1024);
        const uint64_t gv = (uint64_t)(V  + (size_t)t0 * 1024);
        #pragma unroll
        for (int j = 0; j < 4; ++j) {
            const uint32_t voff = (uint32_t)(tid + j * 256) * 16u; // byte offset in tile
            const uint32_t ldq = voff;
            const uint32_t ldk = 16384u + voff;
            const uint32_t ldv = 32768u + voff;
            asm volatile("global_load_async_to_lds_b128 %0, %1, %2"
                         :: "v"(ldq), "v"(voff), "s"(gq) : "memory");
            asm volatile("global_load_async_to_lds_b128 %0, %1, %2"
                         :: "v"(ldk), "v"(voff), "s"(gk) : "memory");
            asm volatile("global_load_async_to_lds_b128 %0, %1, %2"
                         :: "v"(ldv), "v"(voff), "s"(gv) : "memory");
        }
        asm volatile("s_wait_asynccnt 0x0" ::: "memory");
    }
    __syncthreads();

    const _Float16* sQ = smem;
    const _Float16* sK = smem + 8192;
    const _Float16* sV = smem + 16384;
    const int base = w * 1024;          // this wave's token in LDS (halfs)

    // A fragments: q rows (heads), K = dh (0..31)
    v16h aq[2];
    #pragma unroll
    for (int mt = 0; mt < 2; ++mt) {
        const int h = mt * 16 + lr;
        const uint32_t* src = (const uint32_t*)&sQ[base + h * 32];
        uint32_t* ap = (uint32_t*)&aq[mt];
        #pragma unroll
        for (int p = 0; p < 8; ++p) {
            const int kk = (p < 4) ? (hi * 8 + 2 * p) : (16 + hi * 8 + 2 * (p - 4));
            ap[p] = src[kk >> 1];
        }
    }
    // B fragments: B[K=d][N=t] = k[t][d]
    v16h bk[2];
    #pragma unroll
    for (int nt = 0; nt < 2; ++nt) {
        const int t = nt * 16 + lr;
        const uint32_t* src = (const uint32_t*)&sK[base + t * 32];
        uint32_t* bp = (uint32_t*)&bk[nt];
        #pragma unroll
        for (int p = 0; p < 8; ++p)
            bp[p] = src[((p < 4) ? (hi * 8 + 2 * p) : (16 + hi * 8 + 2 * (p - 4))) >> 1];
    }

    v8f sc[2][2] = {};
    #pragma unroll
    for (int mt = 0; mt < 2; ++mt)
        #pragma unroll
        for (int nt = 0; nt < 2; ++nt)
            sc[mt][nt] = __builtin_amdgcn_wmma_f32_16x16x32_f16(
                false, aq[mt], false, bk[nt], (short)0, sc[mt][nt], false, false);

    const float SCALE = 0.03125f;       // 1/sqrt(1024)
    #pragma unroll
    for (int mt = 0; mt < 2; ++mt) {
        #pragma unroll
        for (int i = 0; i < 8; ++i) {
            float v0 = sc[mt][0][i] * SCALE;     // t = lr
            float v1 = sc[mt][1][i] * SCALE;     // t = 16 + lr
            float mx = fmaxf(v0, v1);
            #pragma unroll
            for (int off = 1; off < 16; off <<= 1)
                mx = fmaxf(mx, __shfl_xor(mx, off, 32));
            const float e0 = __expf(v0 - mx);
            const float e1 = __expf(v1 - mx);
            float sm = e0 + e1;
            #pragma unroll
            for (int off = 1; off < 16; off <<= 1)
                sm += __shfl_xor(sm, off, 32);
            const float inv = 1.0f / sm;

            const int h = mt * 16 + hi * 8 + i;  // row of scores
            const _Float16* vrow = &sV[base + h * 32];
            float* orow = Out + (size_t)(t0 + w) * 1024 + h * 32;
            orow[lr]      = e0 * inv * (float)vrow[lr];
            orow[16 + lr] = e1 * inv * (float)vrow[16 + lr];
        }
    }
}

// ---------------------------------------------------------------------------
extern "C" void kernel_launch(void* const* d_in, const int* in_sizes, int n_in,
                              void* d_out, int out_size, void* d_ws, size_t ws_size,
                              hipStream_t stream)
{
    const float* query = (const float*)d_in[0];
    const float* key   = (const float*)d_in[1];
    const float* value = (const float*)d_in[2];
    const float* Wq    = (const float*)d_in[3];
    const float* bq    = (const float*)d_in[4];
    const float* Wk    = (const float*)d_in[5];
    const float* bk    = (const float*)d_in[6];
    const float* Wv    = (const float*)d_in[7];
    const float* bv    = (const float*)d_in[8];
    float* out = (float*)d_out;

    const int Mtot = 4 * 8192;   // B*S = 32768 tokens
    const int N = 1024, K = 1024;

    // workspace: q,k,v projections in fp16 (3 * 64 MB = 192 MB)
    _Float16* qp = (_Float16*)d_ws;
    _Float16* kp = qp + (size_t)Mtot * N;
    _Float16* vp = kp + (size_t)Mtot * N;

    dim3 g1(N / BN, Mtot / BM);  // (8, 256): N-blocks adjacent -> A panel L2 reuse
    qkv_gemm<<<g1, 256, 0, stream>>>(query, Wq, bq, qp, Mtot, N, K);
    qkv_gemm<<<g1, 256, 0, stream>>>(key,   Wk, bk, kp, Mtot, N, K);
    qkv_gemm<<<g1, 256, 0, stream>>>(value, Wv, bv, vp, Mtot, N, K);

    attn_stage2<<<Mtot / 8, 256, 49152, stream>>>(qp, kp, vp, out);
}
